// SpikingHybridCoreFlow_61589831025170
// MI455X (gfx1250) — matrix operand order repaired and checked by reading, not compiled, not run on hardware
//
#include <hip/hip_runtime.h>
#include <stdint.h>

// ---------------- types ----------------
typedef __bf16 bf16_t;
typedef __attribute__((ext_vector_type(16))) __bf16       v16bf;
typedef __attribute__((ext_vector_type(8)))  float        v8f;
typedef __attribute__((ext_vector_type(8)))  unsigned int v8u;
typedef __attribute__((ext_vector_type(4)))  int          v4i;

#define N_DIM 2048
#define B_DIM 64
#define C_DIM 4
#define T_SIM 16
#define KCHUNK 128   // K elements staged in LDS per block per step
#define GEMM_BLOCKS 64
#define SPK_BLOCKS  16

// CDNA5 async global->LDS path, if the toolchain exposes it
#if defined(__has_builtin)
#if __has_builtin(__builtin_amdgcn_global_load_async_to_lds_b128) && \
    __has_builtin(__builtin_amdgcn_s_wait_asynccnt)
#define HAVE_ASYNC_LDS 1
#endif
#endif

#if defined(HAVE_ASYNC_LDS)
// builtin expects (global int4*, lds int4*, imm offset, imm cpol)
typedef __attribute__((address_space(1))) v4i v4i_glb;
typedef __attribute__((address_space(3))) v4i v4i_lds;
#define ASYNC_WAIT() __builtin_amdgcn_s_wait_asynccnt(0)
#else
#define ASYNC_WAIT()
#endif

// pack two 16-byte loads (8 dwords = 16 bf16) into one WMMA operand
__device__ __forceinline__ v16bf pack_op(uint4 lo, uint4 hi) {
    v8u t;
    t[0] = lo.x; t[1] = lo.y; t[2] = lo.z; t[3] = lo.w;
    t[4] = hi.x; t[5] = hi.y; t[6] = hi.z; t[7] = hi.w;
    return __builtin_bit_cast(v16bf, t);
}

// load one B operand (32x16 bf16) from a transposed-weight row: two 16B loads
__device__ __forceinline__ v16bf load_B(const bf16_t* row, int k0) {
    uint4 b0 = *(const uint4*)(row + k0);
    uint4 b1 = *(const uint4*)(row + k0 + 8);
    return pack_op(b0, b1);
}

// in_spikes value for one element at a given cycle (matches _uniform_spikes)
__device__ __forceinline__ float spike_val(float xv, int cycle) {
    float n_spk = rintf(xv * (float)T_SIM);              // round-half-even like jnp.round
    float r = 0.0f;
    if (cycle < T_SIM) {
        bool mask = (n_spk != 0.0f) && (n_spk != (float)T_SIM);
        float n_safe  = fmaxf(n_spk, 1.0f);
        float spacing = (float)T_SIM / n_safe;
        float cy = (float)cycle;
        bool res = mask && (floorf(cy / spacing) < n_safe) &&
                   (floorf(fmodf(cy, spacing)) == 0.0f);
        r = res ? 1.0f : 0.0f;
        if (n_spk == (float)T_SIM) r = 1.0f;
    }
    return r;
}

// ---------------- weight prep: fp32 -> (hi,lo) bf16, transposed to [c][n][a] ----------------
__global__ __launch_bounds__(256) void prep_weights(const float* __restrict__ w,
                                                    bf16_t* __restrict__ wt_hi,
                                                    bf16_t* __restrict__ wt_lo) {
    size_t i = (size_t)blockIdx.x * 256 + threadIdx.x;   // over 4*2048*2048
    int c = (int)(i >> 22);
    int rem = (int)(i & 4194303);
    int a = rem >> 11;
    int n = rem & 2047;
    float v = w[i];
    bf16_t h = (bf16_t)v;
    bf16_t l = (bf16_t)(v - (float)h);
    size_t o = ((size_t)c << 22) + ((size_t)n << 11) + (size_t)a;
    wt_hi[o] = h;
    wt_lo[o] = l;
}

// ---------------- state init ----------------
__global__ __launch_bounds__(256) void init_state(float* __restrict__ memb,
                                                  bf16_t* __restrict__ sig,
                                                  float* __restrict__ out) {
    int i = blockIdx.x * 256 + threadIdx.x;              // over 4*64*2048 = 524288
    memb[i] = 0.0f;
    sig[i]  = (bf16_t)0.0f;
    if (i < B_DIM * N_DIM) out[i] = 0.0f;
}

// ---------------- standalone input spike generation (cycle 0 prologue) ----------------
__global__ __launch_bounds__(256) void spikegen(const float* __restrict__ x,
                                                bf16_t* __restrict__ sig0,
                                                int cycle) {
    int i = blockIdx.x * 256 + threadIdx.x;              // over 64*2048
    sig0[i] = (bf16_t)spike_val(x[i], cycle);
}

// stage 64 rows x KCHUNK k of the signal plane into LDS (16 KB), 256 threads x 4 x 16B
__device__ __forceinline__ void stage_chunk(const bf16_t* __restrict__ gbase,  // sigBase + kc0
                                            bf16_t* lbuf, int tid) {
#pragma unroll
    for (int j = 0; j < 4; ++j) {
        int s   = tid + 256 * j;          // 0..1023 segments of 16B
        int row = s >> 4;                 // 16 segments per 256B row
        int off = (s & 15) << 4;          // byte offset within row chunk
        const char* g = (const char*)(gbase + (size_t)row * N_DIM) + off;
        char*       l = (char*)lbuf + row * (KCHUNK * 2) + off;
#if defined(HAVE_ASYNC_LDS)
        __builtin_amdgcn_global_load_async_to_lds_b128((v4i_glb*)g, (v4i_lds*)l, 0, 0);
#else
        *(uint4*)l = *(const uint4*)g;
#endif
    }
}

// ---------------- one simulation cycle ----------------
// blocks 0..63 : 4 batched GEMMs + threshold/spike (A tile shared via LDS;
//                wave = one (core, 16-wide N tile), all M=64 via 4 accumulators)
// blocks 64..79: input spikes for cycle+1 -> sig_out plane 0 (disjoint region)
__global__ __launch_bounds__(256) void snn_cycle_kernel(const bf16_t* __restrict__ wt_hi,
                                                        const bf16_t* __restrict__ wt_lo,
                                                        const float*  __restrict__ biases,
                                                        const float*  __restrict__ thresholds,
                                                        float*        __restrict__ memb,
                                                        const bf16_t* __restrict__ sig_in,
                                                        bf16_t*       __restrict__ sig_out,
                                                        float*        __restrict__ counts,
                                                        const float*  __restrict__ x,
                                                        int            cycle) {
    __shared__ __align__(16) bf16_t smem[2][B_DIM * KCHUNK];   // 2 x 16 KB

    const int tid = threadIdx.x;

    if (blockIdx.x >= GEMM_BLOCKS) {
        // ---- fused spike generation for NEXT cycle into sig_out plane 0 ----
        int base = (blockIdx.x - GEMM_BLOCKS) * 256 + tid;     // 16 blocks x 256 threads
        const int stride = SPK_BLOCKS * 256;                   // 4096
#pragma unroll
        for (int j = 0; j < (B_DIM * N_DIM) / (SPK_BLOCKS * 256); ++j) {
            int i = base + j * stride;
            sig_out[i] = (bf16_t)spike_val(x[i], cycle + 1);
        }
        return;
    }

    const int lane = tid & 31;
    const int wid  = blockIdx.x * 8 + (tid >> 5);          // 0..511
    const int c    = wid >> 7;                             // core 0..3 (uniform per block)
    const int n0   = (wid & 127) << 4;                     // output N tile base
    const int nl   = lane & 15;
    const int half = lane >> 4;                            // lane half select

    const bf16_t* sigBase = sig_in + ((size_t)c << 17);    // c*64*2048

    // B operand rows: transposed weights wt[c][n][a]; lane nl -> column n0+nl
    const bf16_t* brh = wt_hi + ((size_t)c << 22) + ((size_t)(n0 + nl) << 11) + (half << 4);
    const bf16_t* brl = wt_lo + ((size_t)c << 22) + ((size_t)(n0 + nl) << 11) + (half << 4);

    v8f acc0 = {}; v8f acc1 = {}; v8f acc2 = {}; v8f acc3 = {};

    // prologue: stage first chunk, prefetch first B pair into registers
    stage_chunk(sigBase, &smem[0][0], tid);
    v16bf Bh_n = load_B(brh, 0);
    v16bf Bl_n = load_B(brl, 0);
    ASYNC_WAIT();
    __syncthreads();

    int cur = 0;
    for (int kc0 = 0; kc0 < N_DIM; kc0 += KCHUNK) {
        if (kc0 + KCHUNK < N_DIM)
            stage_chunk(sigBase + kc0 + KCHUNK, &smem[cur ^ 1][0], tid);

        const bf16_t* lbase = &smem[cur][0];
#pragma unroll
        for (int rs = 0; rs < KCHUNK / 32; ++rs) {
            const int k0  = kc0 + rs * 32;        // global K base of this WMMA step
            const int rel = rs * 32;              // K base within LDS chunk

            // consume B prefetched last step; prefetch B for next step (one full
            // step of latency cover; final overread lands in adjacent ws, unused)
            v16bf Bh = Bh_n;
            v16bf Bl = Bl_n;
            Bh_n = load_B(brh, k0 + 32);
            Bl_n = load_B(brl, k0 + 32);

            // A operands from LDS: distinct SSA values -> distinct register sets
            const bf16_t* arb = lbase + nl * KCHUNK + rel + (half << 4);
            uint4 a00 = *(const uint4*)(arb + 0 * 16 * KCHUNK);
            uint4 a01 = *(const uint4*)(arb + 0 * 16 * KCHUNK + 8);
            uint4 a10 = *(const uint4*)(arb + 1 * 16 * KCHUNK);
            uint4 a11 = *(const uint4*)(arb + 1 * 16 * KCHUNK + 8);
            uint4 a20 = *(const uint4*)(arb + 2 * 16 * KCHUNK);
            uint4 a21 = *(const uint4*)(arb + 2 * 16 * KCHUNK + 8);
            uint4 a30 = *(const uint4*)(arb + 3 * 16 * KCHUNK);
            uint4 a31 = *(const uint4*)(arb + 3 * 16 * KCHUNK + 8);
            v16bf A0 = pack_op(a00, a01);
            v16bf A1 = pack_op(a10, a11);
            v16bf A2 = pack_op(a20, a21);
            v16bf A3 = pack_op(a30, a31);

            // 8 WMMAs; consecutive ops never share an accumulator
            acc0 = __builtin_amdgcn_wmma_f32_16x16x32_bf16(false, A0, false, Bh, (short)0, acc0, false, false);
            acc1 = __builtin_amdgcn_wmma_f32_16x16x32_bf16(false, A1, false, Bh, (short)0, acc1, false, false);
            acc2 = __builtin_amdgcn_wmma_f32_16x16x32_bf16(false, A2, false, Bh, (short)0, acc2, false, false);
            acc3 = __builtin_amdgcn_wmma_f32_16x16x32_bf16(false, A3, false, Bh, (short)0, acc3, false, false);
            acc0 = __builtin_amdgcn_wmma_f32_16x16x32_bf16(false, A0, false, Bl, (short)0, acc0, false, false);
            acc1 = __builtin_amdgcn_wmma_f32_16x16x32_bf16(false, A1, false, Bl, (short)0, acc1, false, false);
            acc2 = __builtin_amdgcn_wmma_f32_16x16x32_bf16(false, A2, false, Bl, (short)0, acc2, false, false);
            acc3 = __builtin_amdgcn_wmma_f32_16x16x32_bf16(false, A3, false, Bl, (short)0, acc3, false, false);
        }

        ASYNC_WAIT();          // our async copies into the other buffer are done
        __syncthreads();       // everyone finished reading cur / writing next
        cur ^= 1;
    }

    // epilogue: memb update, spike, route spikes to next core / counts
    const float thr_v  = thresholds[c];
    const float bias_v = biases[(c << 11) + n0 + nl];

#define EPILOGUE(MT, ACC)                                                               \
    {                                                                                   \
        _Pragma("unroll")                                                               \
        for (int r = 0; r < 8; ++r) {                                                   \
            int m = (MT)*16 + r + (half << 3);                                          \
            size_t idx = ((size_t)(c * B_DIM + m) << 11) + (size_t)(n0 + nl);           \
            float v = memb[idx] + ACC[r] + bias_v;                                      \
            float sp = (thr_v < v) ? 1.0f : 0.0f;                                       \
            memb[idx] = v - sp * thr_v;                                                 \
            if (c < 3) {                                                                \
                sig_out[((size_t)((c + 1) * B_DIM + m) << 11) + (size_t)(n0 + nl)]      \
                    = (bf16_t)sp;                                                       \
            } else {                                                                    \
                size_t oidx = ((size_t)m << 11) + (size_t)(n0 + nl);                    \
                counts[oidx] += sp;                                                     \
            }                                                                           \
        }                                                                               \
    }
    EPILOGUE(0, acc0)
    EPILOGUE(1, acc1)
    EPILOGUE(2, acc2)
    EPILOGUE(3, acc3)
#undef EPILOGUE
}

// ---------------- finalize: counts / T ----------------
__global__ __launch_bounds__(256) void finalize(float* __restrict__ out) {
    int i = blockIdx.x * 256 + threadIdx.x;              // over 64*2048
    out[i] *= (1.0f / (float)T_SIM);
}

// ---------------- host orchestration ----------------
extern "C" void kernel_launch(void* const* d_in, const int* in_sizes, int n_in,
                              void* d_out, int out_size, void* d_ws, size_t ws_size,
                              hipStream_t stream) {
    (void)in_sizes; (void)n_in; (void)out_size; (void)ws_size;

    const float* x       = (const float*)d_in[0];   // (64, 2048)
    const float* w       = (const float*)d_in[1];   // (4, 2048, 2048)
    const float* biases  = (const float*)d_in[2];   // (4, 2048)
    const float* thresh  = (const float*)d_in[3];   // (4,)
    float*       out     = (float*)d_out;           // (64, 2048) counts/T

    char* ws = (char*)d_ws;
    bf16_t* wt_hi = (bf16_t*)(ws);                          // 32 MB
    bf16_t* wt_lo = (bf16_t*)(ws + ((size_t)32 << 20));     // 32 MB
    float*  memb  = (float*) (ws + ((size_t)64 << 20));     //  2 MB
    bf16_t* sigA  = (bf16_t*)(ws + ((size_t)66 << 20));     //  1 MB (C,B,N bf16)
    bf16_t* sigB  = (bf16_t*)(ws + ((size_t)67 << 20));     //  1 MB

    // split + transpose weights (L2-resident afterwards: 64 MB total)
    prep_weights<<<65536, 256, 0, stream>>>(w, wt_hi, wt_lo);
    // zero membrane, zero signal buffer (all planes), zero counts
    init_state<<<2048, 256, 0, stream>>>(memb, sigA, out);
    // input spikes for cycle 0 into plane 0
    spikegen<<<512, 256, 0, stream>>>(x, sigA, 0);

    const int cycles = C_DIM + T_SIM;   // 20
    for (int t = 0; t < cycles; ++t) {
        // GEMM blocks consume sigA, produce sigB planes 1..3 + memb/counts;
        // fused spikegen blocks produce sigB plane 0 for cycle t+1
        snn_cycle_kernel<<<GEMM_BLOCKS + SPK_BLOCKS, 256, 0, stream>>>(
            wt_hi, wt_lo, biases, thresh, memb, sigA, sigB, out, x, t);
        bf16_t* tmp = sigA; sigA = sigB; sigB = tmp;
    }

    finalize<<<512, 256, 0, stream>>>(out);
}
